// GAM0_70918499992173
// MI455X (gfx1250) — compile-verified, matrix operand order checked by reading, not run on hardware
//
#include <hip/hip_runtime.h>
#include <hip/hip_bf16.h>

typedef __attribute__((ext_vector_type(16))) __bf16 v16bf;
typedef __attribute__((ext_vector_type(8)))  float  v8f;
typedef __attribute__((ext_vector_type(2)))  float  v2f;

#define HH 256
#define WW 256
#define HW (HH * WW)
#define SXC 72   // padded channel stride of x slab (16B-aligned 8-ch runs, conflict-friendly)
#define S12C 20  // padded channel stride of x12 slab

// jnp 'reflect' padding for +/-1 offsets: -1 -> 1, N -> N-2
__device__ __forceinline__ int refl(int i, int n) {
  return (i < 0) ? -i : ((i >= n) ? (2 * n - 2 - i) : i);
}
// float -> bf16 bits via native convert (v_cvt_*bf16_f32)
__device__ __forceinline__ unsigned short bfbits(float f) {
  return __builtin_bit_cast(unsigned short, static_cast<__bf16>(f));
}
// exact bf16-pair -> float2 (one shl + one and)
__device__ __forceinline__ v2f unpk(unsigned u) {
  v2f r;
  r.x = __uint_as_float(u << 16);
  r.y = __uint_as_float(u & 0xffff0000u);
  return r;
}
// two 16B LDS loads -> one WMMA 16x32 bf16 operand
__device__ __forceinline__ v16bf combine(uint4 a, uint4 b) {
  union { uint4 q[2]; v16bf v; } u;
  u.q[0] = a; u.q[1] = b;
  return u.v;
}

// ---------------------------------------------------------------------------
// Fully fused SAN block. One 256-thread block per (n, h) output row.
//  1) stage x[64ch][3 reflected rows][256 cols] -> LDS bf16 (layout [r][w][c])
//  2) x12 = [conv1;conv2](16x64) @ slab via WMMA -> LDS (bf16, [r][w][c])
//  3) per-pixel: 9-offset MLP + softmax (LDS reads, float2/pk-fma math)
//  4) y = sum_k w_k * x[nbr_k] from LDS (pk-fma); out = conv3_w @ y + b3 (WMMA)
// ---------------------------------------------------------------------------
__global__ void __launch_bounds__(256) san_fused(
    const float* __restrict__ x,
    const float* __restrict__ w1, const float* __restrict__ b1,
    const float* __restrict__ w2, const float* __restrict__ b2,
    const float* __restrict__ w3, const float* __restrict__ b3,
    const float* __restrict__ wp,
    const float* __restrict__ bn1g, const float* __restrict__ bn1b,
    const float* __restrict__ bn1m, const float* __restrict__ bn1v,
    const float* __restrict__ wmid,
    const float* __restrict__ bn2g, const float* __restrict__ bn2b,
    const float* __restrict__ bn2m, const float* __restrict__ bn2v,
    const float* __restrict__ wout, const float* __restrict__ bout,
    float* __restrict__ out) {
  __shared__ alignas(16) unsigned short sx[3 * WW * SXC];    // 110592 B
  __shared__ alignas(16) unsigned short s12[3 * WW * S12C];  //  30720 B
  __shared__ alignas(16) unsigned short sw3[64 * 64];        //   8192 B
  __shared__ alignas(16) unsigned short sw12[16 * 64];       //   2048 B
  // sp layout (all float2 reads 8B-aligned):
  // [0..79] w_mid  [80..87] w_out  [88] b_out  [89] pad
  // [90..99] bn1 scale  [100..109] bn1 shift
  // [110..117] bn2 scale  [118..125] bn2 shift  [126..129] convp_w
  __shared__ alignas(16) float sp[132];

  const int t = threadIdx.x;
  const int n = blockIdx.x >> 8;
  const int h = blockIdx.x & 255;
  const float* xn = x + (long)n * 64 * HW;

  const int r0 = refl(h - 1, HH), r2 = refl(h + 1, HH);

  // ---- stage small params (BN folded to scale/shift) ----
  if (t < 80) sp[t] = wmid[t];
  if (t >= 80 && t < 88) sp[t] = wout[t - 80];
  if (t == 88) sp[88] = bout[0];
  if (t < 10) {
    float sc = bn1g[t] * rsqrtf(bn1v[t] + 1e-5f);
    sp[90 + t] = sc;
    sp[100 + t] = bn1b[t] - bn1m[t] * sc;
  }
  if (t >= 16 && t < 24) {
    int c = t - 16;
    float sc = bn2g[c] * rsqrtf(bn2v[c] + 1e-5f);
    sp[110 + c] = sc;
    sp[118 + c] = bn2b[c] - bn2m[c] * sc;
  }
  if (t >= 32 && t < 36) sp[126 + (t - 32)] = wp[t - 32];

  // ---- stage weights as bf16 ----
#pragma unroll
  for (int i = 0; i < 16; ++i) {
    int idx = i * 256 + t;
    sw3[idx] = bfbits(w3[idx]);
  }
#pragma unroll
  for (int i = 0; i < 4; ++i) {
    int idx = i * 256 + t;
    sw12[idx] = bfbits(idx < 512 ? w1[idx] : w2[idx - 512]);
  }

  // ---- stage x slab: 64ch x 3rows x 256cols (no divisions) ----
  {
    const int tq = t >> 6;        // channel sub-index 0..3
    const int tcg = t & 63;       // 4-column group 0..63
#pragma unroll
    for (int cc = 0; cc < 16; ++cc) {
      const int c = cc * 4 + tq;
      const float* src = xn + (long)c * HW + tcg * 4;
#pragma unroll
      for (int r = 0; r < 3; ++r) {
        const int rimg = (r == 0) ? r0 : ((r == 1) ? h : r2);
        const float4 v = *reinterpret_cast<const float4*>(src + rimg * WW);
        unsigned short* dst = sx + (r * WW + tcg * 4) * SXC + c;
        dst[0 * SXC] = bfbits(v.x);
        dst[1 * SXC] = bfbits(v.y);
        dst[2 * SXC] = bfbits(v.z);
        dst[3 * SXC] = bfbits(v.w);
      }
    }
  }
  __syncthreads();

  const int lane = t & 31, wv = t >> 5;
  const int hi = lane >> 4, ln = lane & 15;

  // ---- x12 GEMM over the 768 slab positions (48 N-tiles, 6 per wave) ----
  v16bf A12[2];
#pragma unroll
  for (int s = 0; s < 2; ++s) {
    uint4 qa = *reinterpret_cast<const uint4*>(sw12 + ln * 64 + 32 * s + 8 * hi);
    uint4 qb = *reinterpret_cast<const uint4*>(sw12 + ln * 64 + 32 * s + 16 + 8 * hi);
    A12[s] = combine(qa, qb);
  }
#pragma unroll
  for (int p = 0; p < 6; ++p) {
    int pos = (wv * 6 + p) * 16 + ln;  // slab position 0..767
    v8f acc = {};
#pragma unroll
    for (int s = 0; s < 2; ++s) {
      uint4 qa = *reinterpret_cast<const uint4*>(sx + pos * SXC + 32 * s + 8 * hi);
      uint4 qb = *reinterpret_cast<const uint4*>(sx + pos * SXC + 32 * s + 16 + 8 * hi);
      v16bf B = combine(qa, qb);
      acc = __builtin_amdgcn_wmma_f32_16x16x32_bf16(false, A12[s], false, B, (short)0, acc, false, false);
    }
    // C: VGPR v -> channel v + 8*hi; store bias-added bf16 pairs
#pragma unroll
    for (int j = 0; j < 4; ++j) {
      float bl = hi ? b2[2 * j] : b1[2 * j];
      float bh = hi ? b2[2 * j + 1] : b1[2 * j + 1];
      unsigned lo = bfbits(acc[2 * j] + bl);
      unsigned hw_ = bfbits(acc[2 * j + 1] + bh);
      *reinterpret_cast<unsigned*>(s12 + pos * S12C + 8 * hi + 2 * j) = lo | (hw_ << 16);
    }
  }
  __syncthreads();

  // ---- Phase 1: per-pixel (thread t owns column w=t) attention weights ----
  const int w = t;
  const v2f* s1v = reinterpret_cast<const v2f*>(sp + 90);
  const v2f* t1v = reinterpret_cast<const v2f*>(sp + 100);
  v2f x1v[4];
  {
    uint2 a = *reinterpret_cast<const uint2*>(s12 + (WW + w) * S12C + 0);
    uint2 b = *reinterpret_cast<const uint2*>(s12 + (WW + w) * S12C + 4);
    x1v[0] = unpk(a.x); x1v[1] = unpk(a.y);
    x1v[2] = unpk(b.x); x1v[3] = unpk(b.y);
  }
  float lg[9];
  int nbr[9];
#pragma unroll
  for (int k = 0; k < 9; ++k) {
    const int di = k / 3 - 1, dj = k % 3 - 1;
    const int rq = (di < 0) ? r0 : ((di > 0) ? r2 : h);  // reflected image row
    const int nw = refl(w + dj, WW);
    const int spos = (1 + di) * WW + nw;                 // slab position
    nbr[k] = spos;
    const float dlw = (2.0f / 255.0f) * (float)(w - nw);
    const float dlh = (2.0f / 255.0f) * (float)(h - rq);
    uint2 a = *reinterpret_cast<const uint2*>(s12 + spos * S12C + 8);
    uint2 b = *reinterpret_cast<const uint2*>(s12 + spos * S12C + 12);
    v2f f2[5];
    f2[0] = x1v[0] - unpk(a.x);
    f2[1] = x1v[1] - unpk(a.y);
    f2[2] = x1v[2] - unpk(b.x);
    f2[3] = x1v[3] - unpk(b.y);
    f2[4].x = sp[126] * dlw + sp[127] * dlh;
    f2[4].y = sp[128] * dlw + sp[129] * dlh;
#pragma unroll
    for (int j = 0; j < 5; ++j) {        // BN1 + ReLU (pk fma + scalar max)
      v2f v = f2[j] * s1v[j] + t1v[j];
      f2[j].x = fmaxf(v.x, 0.0f);
      f2[j].y = fmaxf(v.y, 0.0f);
    }
    float z = sp[88];
#pragma unroll
    for (int o = 0; o < 8; ++o) {        // w_mid row dot: 5 pk-fma + h-add
      const v2f* wm = reinterpret_cast<const v2f*>(sp + o * 10);
      v2f acc2 = wm[0] * f2[0];
      acc2 += wm[1] * f2[1];
      acc2 += wm[2] * f2[2];
      acc2 += wm[3] * f2[3];
      acc2 += wm[4] * f2[4];
      float a2 = acc2.x + acc2.y;
      a2 = fmaxf(fmaf(a2, sp[110 + o], sp[118 + o]), 0.0f);  // BN2 + ReLU
      z = fmaf(sp[80 + o], a2, z);
    }
    lg[k] = z;
  }
  float mx = lg[0];
#pragma unroll
  for (int k = 1; k < 9; ++k) mx = fmaxf(mx, lg[k]);
  float sum = 0.0f;
#pragma unroll
  for (int k = 0; k < 9; ++k) { lg[k] = __expf(lg[k] - mx); sum += lg[k]; }
  float inv = 1.0f / sum;
#pragma unroll
  for (int k = 0; k < 9; ++k) lg[k] *= inv;

  // ---- Phase 2: weighted aggregation from LDS (pk-fma) + WMMA conv3 ----
  float* on = out + (long)n * 64 * HW + h * WW;
#pragma unroll
  for (int g = 0; g < 2; ++g) {
    const int src = g * 16 + ln;  // lane's pixel in B layout (within wave)
    float wk[9];
    int nk[9];
#pragma unroll
    for (int k = 0; k < 9; ++k) {
      wk[k] = __shfl(lg[k], src, 32);
      nk[k] = __shfl(nbr[k], src, 32);
    }
    v8f acc0 = {}, acc1 = {}, acc2 = {}, acc3 = {};
#pragma unroll
    for (int s = 0; s < 2; ++s) {
      v2f y2[8];
#pragma unroll
      for (int j = 0; j < 8; ++j) { y2[j].x = 0.0f; y2[j].y = 0.0f; }
#pragma unroll
      for (int k = 0; k < 9; ++k) {
        const uint4 qa = *reinterpret_cast<const uint4*>(sx + nk[k] * SXC + 32 * s + 8 * hi);
        const uint4 qb = *reinterpret_cast<const uint4*>(sx + nk[k] * SXC + 32 * s + 16 + 8 * hi);
        v2f cw; cw.x = wk[k]; cw.y = wk[k];
        y2[0] += cw * unpk(qa.x);
        y2[1] += cw * unpk(qa.y);
        y2[2] += cw * unpk(qa.z);
        y2[3] += cw * unpk(qa.w);
        y2[4] += cw * unpk(qb.x);
        y2[5] += cw * unpk(qb.y);
        y2[6] += cw * unpk(qb.z);
        y2[7] += cw * unpk(qb.w);
      }
      v16bf B;
#pragma unroll
      for (int j = 0; j < 8; ++j) {      // pairs -> v_cvt_pk_bf16_f32
        B[2 * j]     = static_cast<__bf16>(y2[j].x);
        B[2 * j + 1] = static_cast<__bf16>(y2[j].y);
      }
      v16bf A0 = combine(*reinterpret_cast<const uint4*>(sw3 + (0 + ln) * 64 + 32 * s + 8 * hi),
                         *reinterpret_cast<const uint4*>(sw3 + (0 + ln) * 64 + 32 * s + 16 + 8 * hi));
      v16bf A1 = combine(*reinterpret_cast<const uint4*>(sw3 + (16 + ln) * 64 + 32 * s + 8 * hi),
                         *reinterpret_cast<const uint4*>(sw3 + (16 + ln) * 64 + 32 * s + 16 + 8 * hi));
      v16bf A2 = combine(*reinterpret_cast<const uint4*>(sw3 + (32 + ln) * 64 + 32 * s + 8 * hi),
                         *reinterpret_cast<const uint4*>(sw3 + (32 + ln) * 64 + 32 * s + 16 + 8 * hi));
      v16bf A3 = combine(*reinterpret_cast<const uint4*>(sw3 + (48 + ln) * 64 + 32 * s + 8 * hi),
                         *reinterpret_cast<const uint4*>(sw3 + (48 + ln) * 64 + 32 * s + 16 + 8 * hi));
      acc0 = __builtin_amdgcn_wmma_f32_16x16x32_bf16(false, A0, false, B, (short)0, acc0, false, false);
      acc1 = __builtin_amdgcn_wmma_f32_16x16x32_bf16(false, A1, false, B, (short)0, acc1, false, false);
      acc2 = __builtin_amdgcn_wmma_f32_16x16x32_bf16(false, A2, false, B, (short)0, acc2, false, false);
      acc3 = __builtin_amdgcn_wmma_f32_16x16x32_bf16(false, A3, false, B, (short)0, acc3, false, false);
    }
    const int wpix = (wv * 2 + g) * 16 + ln;  // column of this lane's output
#pragma unroll
    for (int v = 0; v < 8; ++v) {
      int r = v + hi * 8;
      on[(long)(0 + r)  * HW + wpix] = acc0[v] + b3[0 + r];
      on[(long)(16 + r) * HW + wpix] = acc1[v] + b3[16 + r];
      on[(long)(32 + r) * HW + wpix] = acc2[v] + b3[32 + r];
      on[(long)(48 + r) * HW + wpix] = acc3[v] + b3[48 + r];
    }
  }
}

extern "C" void kernel_launch(void* const* d_in, const int* in_sizes, int n_in,
                              void* d_out, int out_size, void* d_ws, size_t ws_size,
                              hipStream_t stream) {
  const float* x    = (const float*)d_in[0];
  const float* w1   = (const float*)d_in[1];
  const float* b1   = (const float*)d_in[2];
  const float* w2   = (const float*)d_in[3];
  const float* b2   = (const float*)d_in[4];
  const float* w3   = (const float*)d_in[5];
  const float* b3   = (const float*)d_in[6];
  const float* wp   = (const float*)d_in[7];
  // d_in[8] = convp_b: cancels in p(center) - p(neighbor)
  const float* bn1g = (const float*)d_in[9];
  const float* bn1b = (const float*)d_in[10];
  const float* bn1m = (const float*)d_in[11];
  const float* bn1v = (const float*)d_in[12];
  const float* wmid = (const float*)d_in[13];
  const float* bn2g = (const float*)d_in[14];
  const float* bn2b = (const float*)d_in[15];
  const float* bn2m = (const float*)d_in[16];
  const float* bn2v = (const float*)d_in[17];
  const float* wout = (const float*)d_in[18];
  const float* bout = (const float*)d_in[19];
  float* out = (float*)d_out;
  (void)d_ws; (void)ws_size;  // fully fused: no workspace needed

  dim3 blk(256);               // 8 waves; one (n, h) row per block
  dim3 grd(16 * HH);           // 4096 blocks
  san_fused<<<grd, blk, 0, stream>>>(x, w1, b1, w2, b2, w3, b3, wp,
                                     bn1g, bn1b, bn1m, bn1v, wmid,
                                     bn2g, bn2b, bn2m, bn2v, wout, bout, out);
}
